// WeightlessFFN_10118942949749
// MI455X (gfx1250) — matrix operand order repaired and checked by reading, not compiled, not run on hardware
//
#include <hip/hip_runtime.h>
#include <hip/hip_bf16.h>
#include <stdint.h>

// ---------------------------------------------------------------------------
// Quantized SwiGLU FFN for MI455X (gfx1250, wave32, WMMA + TDM).
//   h   = silu((x @ gate_w^T) * gate_s) * ((x @ up_w^T) * up_s)   [f16 in ws]
//   out = (h @ down_w^T) * down_s                                  [f32]
// int8-range weights are exact in f16 -> V_WMMA_F32_16X16X32_F16.
// Grid order: M-tiles on blockIdx.x so x / h stay resident in the 192MB L2
// while int32 weights stream through once (non-temporal loads).
// Kernel 2 stages its f16 A-tile with the Tensor Data Mover (TDM), using the
// D# pad feature to reproduce the 40-half LDS pitch expected by the fragment
// loads; completion is tracked with TENSORcnt.
// ---------------------------------------------------------------------------

typedef _Float16 f16;
typedef __attribute__((ext_vector_type(16))) _Float16 v16h;
typedef __attribute__((ext_vector_type(8)))  float    v8f;
typedef __attribute__((ext_vector_type(4)))  int      v4i;
typedef __attribute__((ext_vector_type(4)))  float    v4f;
typedef __attribute__((ext_vector_type(4)))  unsigned v4u;
typedef __attribute__((ext_vector_type(8)))  int      v8i;

#define DMODEL 4096
#define DFF    11008
#define NTOK   4096      // 2 * 2048 tokens
#define BM     128       // block tile M
#define BN     64        // block tile N
#define KT     32        // K step (WMMA K)
#define LDP    20        // LDS row pitch in u32 (= 40 halves: 32 + 8 pad)

#if defined(__HIP_DEVICE_COMPILE__) && __has_builtin(__builtin_amdgcn_tensor_load_to_lds)
#define HAVE_TDM 1
#else
#define HAVE_TDM 0
#endif

union FragU { v16h v; unsigned u[8]; };

// A fragment (16-bit A 16x32): lanes 0-15 -> K0..15, lanes 16-31 -> K16..31
__device__ __forceinline__ v16h ldA(const unsigned* lds, int row, int lane) {
  const unsigned* p = lds + row * LDP + ((lane & 16) ? 8 : 0);
  FragU f;
#pragma unroll
  for (int i = 0; i < 8; ++i) f.u[i] = p[i];
  return f.v;
}

// B fragment (16-bit B 32x16): lane&15 = N col; lanes 0-15 -> K0..15, 16-31 -> K16..31
__device__ __forceinline__ v16h ldB(const unsigned* lds, int row, int lane) {
  const unsigned* p = lds + row * LDP + ((lane & 16) ? 8 : 0);
  FragU f;
#pragma unroll
  for (int i = 0; i < 8; ++i) f.u[i] = p[i];
  return f.v;
}

__device__ __forceinline__ v8f wmma_f16(v16h a, v16h b, v8f c) {
  return __builtin_amdgcn_wmma_f32_16x16x32_f16(false, a, false, b, (short)0, c,
                                                false, false);
}

// ---------------------------------------------------------------------------
// Kernel 1: fused gate/up projections + SiLU epilogue, h written as f16.
// ---------------------------------------------------------------------------
__global__ __launch_bounds__(256) void ffn_gateup_kernel(
    const float* __restrict__ x, const int* __restrict__ gw,
    const float* __restrict__ gs, const int* __restrict__ uw,
    const float* __restrict__ us, f16* __restrict__ hout) {
  __shared__ unsigned sA[BM * LDP];
  __shared__ unsigned sG[BN * LDP];
  __shared__ unsigned sU[BN * LDP];

  const int t    = threadIdx.x;
  const int lane = t & 31;
  const int w    = t >> 5;
  const int wm   = w & 3;   // 4 waves over M (4*32 = 128)
  const int wn   = w >> 2;  // 2 waves over N (2*32 = 64)
  const int m0   = blockIdx.x * BM;   // M fastest -> weights reused via L2
  const int n0   = blockIdx.y * BN;

  const int c4 = t & 7;   // vec4 column within K-tile (8 * 4 = 32)
  const int r  = t >> 3;  // 0..31

  v8f ag[2][2], au[2][2];
#pragma unroll
  for (int i = 0; i < 2; ++i)
#pragma unroll
    for (int j = 0; j < 2; ++j) { ag[i][j] = (v8f)(0.0f); au[i][j] = (v8f)(0.0f); }

  for (int k0 = 0; k0 < DMODEL; k0 += KT) {
    // ---- stage x tile (fp32 -> f16), 128 x 32 ----
#pragma unroll
    for (int i = 0; i < 4; ++i) {
      int row = r + 32 * i;
      v4f f = *(const v4f*)(x + (size_t)(m0 + row) * DMODEL + k0 + c4 * 4);
      union { f16 h[4]; unsigned u[2]; } pk;
      pk.h[0] = (f16)f.x; pk.h[1] = (f16)f.y; pk.h[2] = (f16)f.z; pk.h[3] = (f16)f.w;
      sA[row * LDP + c4 * 2 + 0] = pk.u[0];
      sA[row * LDP + c4 * 2 + 1] = pk.u[1];
    }
    // ---- stage gate/up weight tiles (int32 -> f16), 64 x 32 each ----
#pragma unroll
    for (int i = 0; i < 2; ++i) {
      int row = r + 32 * i;
      const v4i* gp = (const v4i*)(gw + (size_t)(n0 + row) * DMODEL + k0 + c4 * 4);
      const v4i* up = (const v4i*)(uw + (size_t)(n0 + row) * DMODEL + k0 + c4 * 4);
      v4i g = __builtin_nontemporal_load(gp);   // stream weights, keep x in L2
      v4i u = __builtin_nontemporal_load(up);
      if (k0 + KT < DMODEL) {                   // global_prefetch next K tile
        __builtin_prefetch((const char*)(gp) + KT * 4, 0, 0);
        __builtin_prefetch((const char*)(up) + KT * 4, 0, 0);
      }
      union { f16 h[4]; unsigned u[2]; } pg, pu;
      pg.h[0] = (f16)(float)g.x; pg.h[1] = (f16)(float)g.y;
      pg.h[2] = (f16)(float)g.z; pg.h[3] = (f16)(float)g.w;
      pu.h[0] = (f16)(float)u.x; pu.h[1] = (f16)(float)u.y;
      pu.h[2] = (f16)(float)u.z; pu.h[3] = (f16)(float)u.w;
      sG[row * LDP + c4 * 2 + 0] = pg.u[0]; sG[row * LDP + c4 * 2 + 1] = pg.u[1];
      sU[row * LDP + c4 * 2 + 0] = pu.u[0]; sU[row * LDP + c4 * 2 + 1] = pu.u[1];
    }
    __syncthreads();

    const int lr = lane & 15;
    v16h a0 = ldA(sA, wm * 32 + lr, lane);
    v16h a1 = ldA(sA, wm * 32 + 16 + lr, lane);
    v16h g0 = ldB(sG, wn * 32 + lr, lane);
    v16h g1 = ldB(sG, wn * 32 + 16 + lr, lane);
    v16h u0 = ldB(sU, wn * 32 + lr, lane);
    v16h u1 = ldB(sU, wn * 32 + 16 + lr, lane);

    ag[0][0] = wmma_f16(a0, g0, ag[0][0]);
    ag[0][1] = wmma_f16(a0, g1, ag[0][1]);
    ag[1][0] = wmma_f16(a1, g0, ag[1][0]);
    ag[1][1] = wmma_f16(a1, g1, ag[1][1]);
    au[0][0] = wmma_f16(a0, u0, au[0][0]);
    au[0][1] = wmma_f16(a0, u1, au[0][1]);
    au[1][0] = wmma_f16(a1, u0, au[1][0]);
    au[1][1] = wmma_f16(a1, u1, au[1][1]);
    __syncthreads();
  }

  // ---- epilogue: dequant, SiLU, multiply, store f16 ----
#pragma unroll
  for (int nt = 0; nt < 2; ++nt) {
    const int n  = n0 + wn * 32 + nt * 16 + (lane & 15);
    const float gsv = gs[n];
    const float usv = us[n];
#pragma unroll
    for (int mt = 0; mt < 2; ++mt) {
#pragma unroll
      for (int rr = 0; rr < 8; ++rr) {
        int m = m0 + wm * 32 + mt * 16 + rr + ((lane & 16) ? 8 : 0);
        float g = ag[mt][nt][rr] * gsv;
        float u = au[mt][nt][rr] * usv;
        float hv = (g / (1.0f + __expf(-g))) * u;  // silu(g) * u
        hout[(size_t)m * DFF + n] = (f16)hv;
      }
    }
  }
}

// ---------------------------------------------------------------------------
// Kernel 2: down projection, h (f16) @ down_w^T * down_s -> f32 out.
// A-tile staged by the Tensor Data Mover (TENSORcnt-tracked DMA) when the
// toolchain exposes the builtin; per-thread fallback otherwise.
// ---------------------------------------------------------------------------
__global__ __launch_bounds__(256) void ffn_down_kernel(
    const f16* __restrict__ h, const int* __restrict__ dw,
    const float* __restrict__ dsc, float* __restrict__ out) {
  __shared__ unsigned sA[BM * LDP];
  __shared__ unsigned sB[BN * LDP];

  const int t    = threadIdx.x;
  const int lane = t & 31;
  const int w    = t >> 5;
  const int wm   = w & 3;
  const int wn   = w >> 2;
  const int m0   = blockIdx.x * BM;   // M fastest -> h stays in L2
  const int n0   = blockIdx.y * BN;

  const int c4 = t & 7;
  const int r4 = t >> 3;  // 0..31
#if !HAVE_TDM
  const int c8 = t & 3;   // 8-half group within K-tile (4 * 8 = 32)
  const int r8 = t >> 2;  // 0..63
#endif

  v8f acc[2][2];
#pragma unroll
  for (int i = 0; i < 2; ++i)
#pragma unroll
    for (int j = 0; j < 2; ++j) acc[i][j] = (v8f)(0.0f);

#if HAVE_TDM
  // D# group 1 (static part of the descriptor):
  //   data_size=2B (code 1), pad_enable, pad_interval: 16 DWORDs (code 3),
  //   pad_amount: 4 DWORDs (code 3)  -> 64B data + 16B pad == 40-half pitch.
  //   tensor_dim0 = DFF, tensor_dim1 = NTOK, tile = 32 x 128, stride0 = DFF.
  const v8i tdm_g1 = {
      (int)((1u << 16) | (1u << 20) | (3u << 22) | (3u << 25)),  // flags
      (int)((DFF & 0xFFFF) << 16),        // tensor_dim0[15:0] in bits[63:48]
      (int)((NTOK & 0xFFFF) << 16),       // tensor_dim0 hi=0, tensor_dim1 lo
      (int)(KT << 16),                    // tensor_dim1 hi=0, tile_dim0=32
      (int)BM,                            // tile_dim1=128, tile_dim2=0
      (int)DFF,                           // tensor_dim0_stride lo
      0, 0                                // stride hi, tensor_dim1_stride
  };
  const unsigned ldsA_addr = (unsigned)(uintptr_t)(&sA[0]);
#endif

  for (int k0 = 0; k0 < DFF; k0 += KT) {
#if HAVE_TDM
    // ---- stage h tile 128x32 f16 via Tensor Data Mover (wave 0 issues) ----
    if (w == 0) {
      unsigned long long ga = (unsigned long long)(uintptr_t)h +
                              ((unsigned long long)m0 * DFF + (unsigned)k0) * 2ull;
      v4u g0 = { 1u,                                   // count=1, user mode
                 ldsA_addr,                            // lds_addr (bytes)
                 (unsigned)ga,                         // global_addr[31:0]
                 (unsigned)((ga >> 32) & 0x01FFFFFFu) | (2u << 30) };  // +type=2
      // 6-arg toolchain variant: (g0, g1, g2, g3, g4, cpol)
      __builtin_amdgcn_tensor_load_to_lds(g0, tdm_g1, (v4i)0, (v4i)0, (v8i)0, 0);
    }
#else
    // ---- stage h tile (already f16), 128 x 32, per-thread fallback ----
#pragma unroll
    for (int i = 0; i < 2; ++i) {
      int row = r8 + 64 * i;
      v4u d = *(const v4u*)(h + (size_t)(m0 + row) * DFF + k0 + c8 * 8);
      sA[row * LDP + c8 * 4 + 0] = d.x;
      sA[row * LDP + c8 * 4 + 1] = d.y;
      sA[row * LDP + c8 * 4 + 2] = d.z;
      sA[row * LDP + c8 * 4 + 3] = d.w;
    }
#endif
    // ---- stage down_w tile (int32 -> f16), 64 x 32 ----
#pragma unroll
    for (int i = 0; i < 2; ++i) {
      int row = r4 + 32 * i;
      const v4i* dp = (const v4i*)(dw + (size_t)(n0 + row) * DFF + k0 + c4 * 4);
      v4i d = __builtin_nontemporal_load(dp);
      if (k0 + KT < DFF) __builtin_prefetch((const char*)(dp) + KT * 4, 0, 0);
      union { f16 h[4]; unsigned u[2]; } pk;
      pk.h[0] = (f16)(float)d.x; pk.h[1] = (f16)(float)d.y;
      pk.h[2] = (f16)(float)d.z; pk.h[3] = (f16)(float)d.w;
      sB[row * LDP + c4 * 2 + 0] = pk.u[0];
      sB[row * LDP + c4 * 2 + 1] = pk.u[1];
    }
#if HAVE_TDM
    if (w == 0) __builtin_amdgcn_s_wait_tensorcnt(0);  // DMA done before barrier
#endif
    __syncthreads();

    const int lr = lane & 15;
    v16h a0 = ldA(sA, wm * 32 + lr, lane);
    v16h a1 = ldA(sA, wm * 32 + 16 + lr, lane);
    v16h b0 = ldB(sB, wn * 32 + lr, lane);
    v16h b1 = ldB(sB, wn * 32 + 16 + lr, lane);

    acc[0][0] = wmma_f16(a0, b0, acc[0][0]);
    acc[0][1] = wmma_f16(a0, b1, acc[0][1]);
    acc[1][0] = wmma_f16(a1, b0, acc[1][0]);
    acc[1][1] = wmma_f16(a1, b1, acc[1][1]);
    __syncthreads();
  }

#pragma unroll
  for (int nt = 0; nt < 2; ++nt) {
    const int n = n0 + wn * 32 + nt * 16 + (lane & 15);
    const float sv = dsc[n];
#pragma unroll
    for (int mt = 0; mt < 2; ++mt) {
#pragma unroll
      for (int rr = 0; rr < 8; ++rr) {
        int m = m0 + wm * 32 + mt * 16 + rr + ((lane & 16) ? 8 : 0);
        out[(size_t)m * DMODEL + n] = acc[mt][nt][rr] * sv;
      }
    }
  }
}

// ---------------------------------------------------------------------------
extern "C" void kernel_launch(void* const* d_in, const int* in_sizes, int n_in,
                              void* d_out, int out_size, void* d_ws, size_t ws_size,
                              hipStream_t stream) {
  (void)in_sizes; (void)n_in; (void)out_size; (void)ws_size;
  const float* x   = (const float*)d_in[0];
  const int*   gw  = (const int*)d_in[1];
  const float* gsc = (const float*)d_in[2];
  const int*   uw  = (const int*)d_in[3];
  const float* usc = (const float*)d_in[4];
  const int*   dw  = (const int*)d_in[5];
  const float* dsc = (const float*)d_in[6];

  f16* hbuf = (f16*)d_ws;  // NTOK * DFF f16 intermediate (~90 MB)

  dim3 blk(256);
  // blockIdx.x = M tiles (fast-varying) so x / h stay resident in 192MB L2
  // while weight stripes stream through HBM exactly once.
  ffn_gateup_kernel<<<dim3(NTOK / BM, DFF / BN), blk, 0, stream>>>(
      x, gw, gsc, uw, usc, hbuf);
  ffn_down_kernel<<<dim3(NTOK / BM, DMODEL / BN), blk, 0, stream>>>(
      hbuf, dw, dsc, (float*)d_out);
}
